// SemanticGraph_49589692400380
// MI455X (gfx1250) — matrix operand order repaired and checked by reading, not compiled, not run on hardware
//
#include <hip/hip_runtime.h>
#include <math.h>

typedef __attribute__((ext_vector_type(8)))  float  v8f;
typedef __attribute__((ext_vector_type(4)))  __bf16 v4bf;
typedef __attribute__((ext_vector_type(8)))  __bf16 v8bf;
typedef __attribute__((ext_vector_type(16))) __bf16 v16bf;

#define NROW 8192
#define DIN  512
#define DH   512

#define BM 128
#define BN 128
#define BK 32
#define LDSS 40   // padded LDS row stride (bf16 elems) to break bank alignment

// ---------------------------------------------------------------------------
// f32 -> bf16 elementwise convert (vectorized x4)
// ---------------------------------------------------------------------------
__global__ __launch_bounds__(256)
void conv_f32_bf16(const float* __restrict__ in, __bf16* __restrict__ out, int n4) {
    int i = blockIdx.x * blockDim.x + threadIdx.x;
    if (i < n4) {
        float4 f = ((const float4*)in)[i];
        v4bf o;
        o[0] = (__bf16)f.x; o[1] = (__bf16)f.y;
        o[2] = (__bf16)f.z; o[3] = (__bf16)f.w;
        ((v4bf*)out)[i] = o;
    }
}

// ---------------------------------------------------------------------------
// Unified bf16 NT GEMM: C[M,Nc] = scale * (A @ B^T) (+bias) (relu) (+addsrc)
//   A: [M,K] bf16 row-major, B: [Nc,K] bf16 row-major
//   outputs: optional f32 C, optional bf16 C (normal or transposed [Nc,M])
// 256 threads = 8 waves; wave grid 2(M) x 4(N); wave tile 64x32 = 8 WMMAs.
// Register-double-buffered: iteration k issues k+1's global loads before the
// WMMA block so s_wait_loadcnt lands after the matrix math.
// ---------------------------------------------------------------------------
__global__ __launch_bounds__(256)
void gemm_bf16_nt(const __bf16* __restrict__ A, const __bf16* __restrict__ B,
                  const float* __restrict__ bias, float scale, int do_relu,
                  const float* __restrict__ addsrc,
                  float* __restrict__ outF, __bf16* __restrict__ outB, int outB_trans,
                  int M, int Nc, int K) {
    __shared__ __bf16 lA[BM * LDSS];
    __shared__ __bf16 lB[BN * LDSS];

    const int tid   = threadIdx.x;
    const int bx    = blockIdx.x, by = blockIdx.y;
    const int wave  = tid >> 5;
    const int lane  = tid & 31;
    const int lrow  = lane & 15;
    const int lhalf = lane >> 4;
    const int wm0   = (wave >> 2) * 64;  // 2 waves along M
    const int wn0   = (wave & 3) * 32;   // 4 waves along N

    const v8f vzero = {0.f,0.f,0.f,0.f,0.f,0.f,0.f,0.f};
    v8f acc[4][2];
#pragma unroll
    for (int i = 0; i < 4; ++i)
#pragma unroll
        for (int j = 0; j < 2; ++j) acc[i][j] = vzero;

    // global->LDS staging: each thread moves 2 chunks of 8 bf16 per tile side
    const int r0 = tid >> 2;          // rows 0..63 (and +64)
    const int c0 = (tid & 3) * 8;     // k-chunk within BK

    const __bf16* Ag = A + (size_t)(by * BM) * K + c0;
    const __bf16* Bg = B + (size_t)(bx * BN) * K + c0;
    const __bf16* pa0 = Ag + (size_t)(r0)      * K;
    const __bf16* pa1 = Ag + (size_t)(r0 + 64) * K;
    const __bf16* pb0 = Bg + (size_t)(r0)      * K;
    const __bf16* pb1 = Bg + (size_t)(r0 + 64) * K;

    const int kt = K / BK;

    // preload tile 0 into registers
    v8bf ra0 = *(const v8bf*)(pa0);
    v8bf ra1 = *(const v8bf*)(pa1);
    v8bf rb0 = *(const v8bf*)(pb0);
    v8bf rb1 = *(const v8bf*)(pb1);

    for (int kk = 0; kk < kt; ++kk) {
        // stage current tile registers -> LDS
        *(v8bf*)&lA[(r0)      * LDSS + c0] = ra0;
        *(v8bf*)&lA[(r0 + 64) * LDSS + c0] = ra1;
        *(v8bf*)&lB[(r0)      * LDSS + c0] = rb0;
        *(v8bf*)&lB[(r0 + 64) * LDSS + c0] = rb1;
        __syncthreads();

        // issue next tile's global loads early (overlaps with WMMA below)
        if (kk + 1 < kt) {
            const int k1 = (kk + 1) * BK;
            ra0 = *(const v8bf*)(pa0 + k1);
            ra1 = *(const v8bf*)(pa1 + k1);
            rb0 = *(const v8bf*)(pb0 + k1);
            rb1 = *(const v8bf*)(pb1 + k1);
            if (kk + 2 < kt) {  // gfx1250 global_prefetch_b8, near-scope
                __builtin_prefetch(pa0 + k1 + BK, 0, 3);
                __builtin_prefetch(pb0 + k1 + BK, 0, 3);
            }
        }

        // A fragment (16x32 bf16): lane holds row lrow, K at lhalf*8 and 16+lhalf*8
        v16bf af[4];
#pragma unroll
        for (int tm = 0; tm < 4; ++tm) {
            const __bf16* p = &lA[(wm0 + tm * 16 + lrow) * LDSS + lhalf * 8];
            v8bf lo = *(const v8bf*)p;
            v8bf hi = *(const v8bf*)(p + 16);
            af[tm] = __builtin_shufflevector(lo, hi, 0,1,2,3,4,5,6,7,8,9,10,11,12,13,14,15);
        }
        // B fragment (32x16 bf16): lane holds col lrow, K at lhalf*8 and 16+lhalf*8
        v16bf bfr[2];
#pragma unroll
        for (int tn = 0; tn < 2; ++tn) {
            const __bf16* p = &lB[(wn0 + tn * 16 + lrow) * LDSS + lhalf * 8];
            v8bf lo = *(const v8bf*)p;
            v8bf hi = *(const v8bf*)(p + 16);
            bfr[tn] = __builtin_shufflevector(lo, hi, 0,1,2,3,4,5,6,7,8,9,10,11,12,13,14,15);
        }
#pragma unroll
        for (int tm = 0; tm < 4; ++tm)
#pragma unroll
            for (int tn = 0; tn < 2; ++tn)
                acc[tm][tn] = __builtin_amdgcn_wmma_f32_16x16x32_bf16(
                    false, af[tm], false, bfr[tn], (short)0, acc[tm][tn], false, false);
        __syncthreads();
    }

    // epilogue: C layout — lane l, VGPR r -> (M = r + 8*(l>>4), N = l&15)
#pragma unroll
    for (int tm = 0; tm < 4; ++tm) {
        const int rowbase = by * BM + wm0 + tm * 16 + lhalf * 8;
#pragma unroll
        for (int tn = 0; tn < 2; ++tn) {
            const int col = bx * BN + wn0 + tn * 16 + lrow;
            const float bv = bias ? bias[col] : 0.f;
            float vals[8];
#pragma unroll
            for (int r = 0; r < 8; ++r) {
                float v = acc[tm][tn][r] * scale + bv;
                if (do_relu) v = fmaxf(v, 0.f);
                if (addsrc) v += addsrc[(size_t)(rowbase + r) * Nc + col];
                vals[r] = v;
                if (outF) outF[(size_t)(rowbase + r) * Nc + col] = v;
            }
            if (outB) {
                if (outB_trans) {  // outB is [Nc, M]; 8 consecutive rows -> b128 store
                    v8bf pk;
#pragma unroll
                    for (int r = 0; r < 8; ++r) pk[r] = (__bf16)vals[r];
                    *(v8bf*)(outB + (size_t)col * M + rowbase) = pk;
                } else {
#pragma unroll
                    for (int r = 0; r < 8; ++r)
                        outB[(size_t)(rowbase + r) * Nc + col] = (__bf16)vals[r];
                }
            }
        }
    }
}

// ---------------------------------------------------------------------------
// Row softmax fused with post-softmax +cooccur and bf16 convert.
// One 256-thread block per row; row cached in 32KB LDS.
// ---------------------------------------------------------------------------
__global__ __launch_bounds__(256)
void softmax_add_bf16(const float* __restrict__ S, const float* __restrict__ C,
                      __bf16* __restrict__ att, int N) {
    __shared__ float row[NROW];
    __shared__ float red[256];
    const int r   = blockIdx.x;
    const int tid = threadIdx.x;
    const float4* S4 = (const float4*)(S + (size_t)r * N);

    float lmax = -3.402823466e38f;
    for (int i = tid; i < N / 4; i += 256) {
        float4 f = S4[i];
        ((float4*)row)[i] = f;
        lmax = fmaxf(lmax, fmaxf(fmaxf(f.x, f.y), fmaxf(f.z, f.w)));
    }
    red[tid] = lmax; __syncthreads();
    for (int s = 128; s > 0; s >>= 1) {
        if (tid < s) red[tid] = fmaxf(red[tid], red[tid + s]);
        __syncthreads();
    }
    const float m = red[0];
    __syncthreads();

    float lsum = 0.f;
    for (int i = tid; i < N / 4; i += 256) {
        float4 f = ((float4*)row)[i];
        f.x = __expf(f.x - m); f.y = __expf(f.y - m);
        f.z = __expf(f.z - m); f.w = __expf(f.w - m);
        ((float4*)row)[i] = f;
        lsum += f.x + f.y + f.z + f.w;
    }
    red[tid] = lsum; __syncthreads();
    for (int s = 128; s > 0; s >>= 1) {
        if (tid < s) red[tid] += red[tid + s];
        __syncthreads();
    }
    const float inv = 1.0f / red[0];

    const float4* C4 = (const float4*)(C + (size_t)r * N);
    v4bf* A4 = (v4bf*)(att + (size_t)r * N);
    for (int i = tid; i < N / 4; i += 256) {
        float4 e = ((float4*)row)[i];
        float4 c = C4[i];
        v4bf o;
        o[0] = (__bf16)(e.x * inv + c.x);
        o[1] = (__bf16)(e.y * inv + c.y);
        o[2] = (__bf16)(e.z * inv + c.z);
        o[3] = (__bf16)(e.w * inv + c.w);
        A4[i] = o;
    }
}

// ---------------------------------------------------------------------------
extern "C" void kernel_launch(void* const* d_in, const int* in_sizes, int n_in,
                              void* d_out, int out_size, void* d_ws, size_t ws_size,
                              hipStream_t stream) {
    const float* x       = (const float*)d_in[0];
    const float* cooccur = (const float*)d_in[1];
    const float* wq0 = (const float*)d_in[2];  const float* bq0 = (const float*)d_in[3];
    const float* wk0 = (const float*)d_in[4];  const float* bk0 = (const float*)d_in[5];
    const float* wv0 = (const float*)d_in[6];  const float* bv0 = (const float*)d_in[7];
    const float* wrs = (const float*)d_in[8];  const float* brs = (const float*)d_in[9];
    const float* wq1 = (const float*)d_in[10]; const float* bq1 = (const float*)d_in[11];
    const float* wk1 = (const float*)d_in[12]; const float* bk1 = (const float*)d_in[13];
    const float* wv1 = (const float*)d_in[14]; const float* bv1 = (const float*)d_in[15];

    char* ws = (char*)d_ws;
    size_t off = 0;
    auto alloc = [&](size_t bytes) -> void* {
        void* p = ws + off;
        off += (bytes + 255) & ~(size_t)255;
        return p;
    };
    const size_t ND  = (size_t)NROW * DH;      // 4M elems
    const size_t WW  = (size_t)DH * DIN;       // 256K elems
    const size_t NN2 = (size_t)NROW * NROW;    // 64M elems

    __bf16* xb    = (__bf16*)alloc(ND * 2);
    __bf16* wq0b  = (__bf16*)alloc(WW * 2);
    __bf16* wk0b  = (__bf16*)alloc(WW * 2);
    __bf16* wv0b  = (__bf16*)alloc(WW * 2);
    __bf16* wrsb  = (__bf16*)alloc(WW * 2);
    __bf16* wq1b  = (__bf16*)alloc(WW * 2);
    __bf16* wk1b  = (__bf16*)alloc(WW * 2);
    __bf16* wv1b  = (__bf16*)alloc(WW * 2);
    __bf16* qb    = (__bf16*)alloc(ND * 2);
    __bf16* kb    = (__bf16*)alloc(ND * 2);
    __bf16* vTb   = (__bf16*)alloc(ND * 2);    // stored transposed [DH, NROW]
    float*  resid = (float*) alloc(ND * 4);
    float*  h0    = (float*) alloc(ND * 4);
    __bf16* h0b   = (__bf16*)alloc(ND * 2);
    __bf16* att   = (__bf16*)alloc(NN2 * 2);
    float*  S     = (float*) alloc(NN2 * 4);

    const float scale = 1.0f / sqrtf((float)DH);
    dim3 blk(256);

    auto conv = [&](const float* in, __bf16* out, size_t n) {
        int n4 = (int)(n / 4);
        conv_f32_bf16<<<dim3((n4 + 255) / 256), blk, 0, stream>>>(in, out, n4);
    };
    auto gemm = [&](const __bf16* A, const __bf16* B, const float* bias, float sc,
                    int relu, const float* add, float* oF, __bf16* oB, int tr,
                    int M, int Nc, int K) {
        dim3 grd(Nc / BN, M / BM);
        gemm_bf16_nt<<<grd, blk, 0, stream>>>(A, B, bias, sc, relu, add, oF, oB, tr, M, Nc, K);
    };

    // precision conversion
    conv(x, xb, ND);
    conv(wq0, wq0b, WW); conv(wk0, wk0b, WW); conv(wv0, wv0b, WW); conv(wrs, wrsb, WW);
    conv(wq1, wq1b, WW); conv(wk1, wk1b, WW); conv(wv1, wv1b, WW);

    // ---- layer 0 ----
    gemm(xb, wq0b, bq0, 1.f, 0, nullptr, nullptr, qb,  0, NROW, DH, DIN);
    gemm(xb, wk0b, bk0, 1.f, 0, nullptr, nullptr, kb,  0, NROW, DH, DIN);
    gemm(xb, wv0b, bv0, 1.f, 0, nullptr, nullptr, vTb, 1, NROW, DH, DIN);
    gemm(xb, wrsb, brs, 1.f, 0, nullptr, resid, nullptr, 0, NROW, DH, DIN);
    gemm(qb, kb, nullptr, scale, 0, nullptr, S, nullptr, 0, NROW, NROW, DH);
    softmax_add_bf16<<<dim3(NROW), blk, 0, stream>>>(S, cooccur, att, NROW);
    gemm(att, vTb, nullptr, 1.f, 1, resid, h0, h0b, 0, NROW, DH, NROW);

    // ---- layer 1 ----
    gemm(h0b, wq1b, bq1, 1.f, 0, nullptr, nullptr, qb,  0, NROW, DH, DH);
    gemm(h0b, wk1b, bk1, 1.f, 0, nullptr, nullptr, kb,  0, NROW, DH, DH);
    gemm(h0b, wv1b, bv1, 1.f, 0, nullptr, nullptr, vTb, 1, NROW, DH, DH);
    gemm(qb, kb, nullptr, scale, 0, nullptr, S, nullptr, 0, NROW, NROW, DH);
    softmax_add_bf16<<<dim3(NROW), blk, 0, stream>>>(S, cooccur, att, NROW);
    gemm(att, vTb, nullptr, 1.f, 1, h0, (float*)d_out, nullptr, 0, NROW, DH, NROW);
}
